// RGDiscrimininator_32504312496833
// MI455X (gfx1250) — matrix operand order repaired and checked by reading, not compiled, not run on hardware
//
#include <hip/hip_runtime.h>
#include <hip/hip_bf16.h>

// ---------------- types for WMMA ----------------
typedef __attribute__((ext_vector_type(16))) __bf16 v16bf;
typedef __attribute__((ext_vector_type(8)))  float  v8f;

#define N_NODES_C  100000
#define N_EDGES_C  1600000
#define N_GRAPHS_C 256
#define IN_DIM_C   64
#define HID_C      128
#define RELU_COEF_C 0.05f

// GEMM tiling
#define MT 64            // rows of output per block
#define NT 128           // output columns (all kernels have N=128)
#define KMAX 256
#define AS_STRIDE (KMAX + 8)   // bf16 elems; 264*2B row stride -> staggered banks
#define WS_STRIDE 34           // 32 + 2 pad

static __device__ __forceinline__ __bf16 f2bf(float f) {
    unsigned u = __builtin_bit_cast(unsigned, f);
    unsigned r = u + 0x7FFFu + ((u >> 16) & 1u);      // round-to-nearest-even
    unsigned short h = (unsigned short)(r >> 16);
    return __builtin_bit_cast(__bf16, h);
}
static __device__ __forceinline__ float bfu2f(unsigned short h) {
    unsigned u = ((unsigned)h) << 16;
    return __builtin_bit_cast(float, u);
}
static __device__ __forceinline__ float bf2f(__bf16 b) {
    return bfu2f(__builtin_bit_cast(unsigned short, b));
}
static __device__ __forceinline__ float leaky(float v) {
    return v > 0.f ? v : RELU_COEF_C * v;
}

// ---------------- zero fill ----------------
__global__ void zero_f32(float* __restrict__ p, long n) {
    long i = (long)blockIdx.x * blockDim.x + threadIdx.x;
    long stride = (long)gridDim.x * blockDim.x;
    for (; i < n; i += stride) p[i] = 0.f;
}

// ---------------- edge scatter (fp32 source) ----------------
__global__ void scatter_f32(const float* __restrict__ x,
                            const int* __restrict__ src,
                            const int* __restrict__ dst,
                            const float* __restrict__ w,
                            float* __restrict__ agg, int lgdim) {
    const int dim = 1 << lgdim;
    long total = (long)N_EDGES_C << lgdim;
    long i = (long)blockIdx.x * blockDim.x + threadIdx.x;
    long stride = (long)gridDim.x * blockDim.x;
    for (; i < total; i += stride) {
        int e = (int)(i >> lgdim);
        int f = (int)(i & (dim - 1));
        int s = src[e], d = dst[e];
        float v = x[(long)s * dim + f] * w[e];
        atomicAdd(&agg[(long)d * dim + f], v);
    }
}

// ---------------- edge scatter (bf16 source) ----------------
__global__ void scatter_bf16(const unsigned short* __restrict__ x,
                             const int* __restrict__ src,
                             const int* __restrict__ dst,
                             const float* __restrict__ w,
                             float* __restrict__ agg, int lgdim) {
    const int dim = 1 << lgdim;
    long total = (long)N_EDGES_C << lgdim;
    long i = (long)blockIdx.x * blockDim.x + threadIdx.x;
    long stride = (long)gridDim.x * blockDim.x;
    for (; i < total; i += stride) {
        int e = (int)(i >> lgdim);
        int f = (int)(i & (dim - 1));
        int s = src[e], d = dst[e];
        float v = bfu2f(x[(long)s * dim + f]) * w[e];
        atomicAdd(&agg[(long)d * dim + f], v);
    }
}

// ---------------- WMMA GEMM: D = leaky([A0 || A1] @ [W0;W1] + bias), bf16 out ----
__global__ __launch_bounds__(256)
void gemm_bias_leaky(const float* __restrict__ A0f, const unsigned short* __restrict__ A0b, int K0,
                     const float* __restrict__ A1f, const unsigned short* __restrict__ A1b, int K1,
                     const float* __restrict__ W0, const float* __restrict__ W1, int splitK,
                     const float* __restrict__ bias,
                     unsigned short* __restrict__ D, int M) {
    __shared__ __bf16 As[MT * AS_STRIDE];
    __shared__ __bf16 Ws[NT * WS_STRIDE];

    const int tid  = threadIdx.x;
    const int lane = tid & 31;
    const int wv   = tid >> 5;
    const int msub = wv & 3;        // 4 M sub-tiles of 16
    const int nhalf = wv >> 1 >> 1; // wv>>2: 2 N halves of 64
    const int m0 = blockIdx.x * MT;
    const int Ktot = K0 + K1;

    // Stage A tile (convert to bf16), zero-pad rows >= M
    for (int idx = tid; idx < MT * Ktot; idx += 256) {
        int r = idx / Ktot, k = idx - r * Ktot;
        int grow = m0 + r;
        float v = 0.f;
        if (grow < M) {
            if (k < K0) v = A0b ? bfu2f(A0b[(long)grow * K0 + k]) : A0f[(long)grow * K0 + k];
            else {
                int kk = k - K0;
                v = A1b ? bfu2f(A1b[(long)grow * K1 + kk]) : A1f[(long)grow * K1 + kk];
            }
        }
        As[r * AS_STRIDE + k] = f2bf(v);
    }

    v8f acc[4];
    #pragma unroll
    for (int t = 0; t < 4; t++)
        #pragma unroll
        for (int v = 0; v < 8; v++) acc[t][v] = 0.f;

    const int am = msub * 16 + (lane & 15);
    const int kh = lane >> 4;

    for (int kt = 0; kt < Ktot; kt += 32) {
        __syncthreads();
        // Stage transposed W chunk: Ws[n][kk], row split between W0/W1 at splitK
        for (int idx = tid; idx < NT * 32; idx += 256) {
            int n = idx >> 5, kk = idx & 31;
            int kg = kt + kk;
            const float* Wp = (kg < splitK) ? (W0 + (long)kg * NT) : (W1 + (long)(kg - splitK) * NT);
            Ws[n * WS_STRIDE + kk] = f2bf(Wp[n]);
        }
        __syncthreads();

        // A fragment: 16-bit A 16x32 layout (ISA 7.12.2)
        v16bf a;
        #pragma unroll
        for (int v = 0; v < 8; v++) {
            int kb = (v < 4 ? 2 * v : 16 + 2 * (v - 4)) + 8 * kh;
            a[2 * v]     = As[am * AS_STRIDE + kt + kb];
            a[2 * v + 1] = As[am * AS_STRIDE + kt + kb + 1];
        }
        #pragma unroll
        for (int t = 0; t < 4; t++) {
            int nn = nhalf * 64 + t * 16 + (lane & 15);
            v16bf b;   // B 32x16: lanes 0-15 -> K 0..7,16..23, lanes 16-31 -> K 8..15,24..31
            #pragma unroll
            for (int v = 0; v < 8; v++) {
                int kk = (v < 4 ? 2 * v : 16 + 2 * (v - 4)) + 8 * kh;
                b[2 * v]     = Ws[nn * WS_STRIDE + kk];
                b[2 * v + 1] = Ws[nn * WS_STRIDE + kk + 1];
            }
            acc[t] = __builtin_amdgcn_wmma_f32_16x16x32_bf16(
                false, a, false, b, (short)0, acc[t], false, false);
        }
    }

    // Epilogue: bias + leaky, store bf16. D layout: m = (lane>>4)*8+v, n = lane&15
    #pragma unroll
    for (int t = 0; t < 4; t++) {
        int n = nhalf * 64 + t * 16 + (lane & 15);
        float bn = bias[n];
        #pragma unroll
        for (int v = 0; v < 8; v++) {
            int m = msub * 16 + (lane >> 4) * 8 + v;
            int grow = m0 + m;
            if (grow < M) {
                float val = leaky(acc[t][v] + bn);
                D[(long)grow * NT + n] = __builtin_bit_cast(unsigned short, f2bf(val));
            }
        }
    }
}

// ---------------- WMMA gate GEMMs + fused sigmoid*tanh + segment-sum pool ----
__global__ __launch_bounds__(256)
void gemm_gate_pool(const unsigned short* __restrict__ Hb, int K0,   // bf16 [M,128]
                    const float* __restrict__ Xf, int K1,            // fp32 [M,64]
                    const float* __restrict__ Wsig, const float* __restrict__ bsig,
                    const float* __restrict__ Wtan, const float* __restrict__ btan,
                    const int* __restrict__ batch,
                    float* __restrict__ pooled, int M) {
    __shared__ __bf16 As[MT * AS_STRIDE];
    __shared__ __bf16 Wss[NT * WS_STRIDE];
    __shared__ __bf16 Wts[NT * WS_STRIDE];

    const int tid  = threadIdx.x;
    const int lane = tid & 31;
    const int wv   = tid >> 5;
    const int msub = wv & 3;
    const int nhalf = wv >> 2;
    const int m0 = blockIdx.x * MT;
    const int Ktot = K0 + K1;   // 192

    for (int idx = tid; idx < MT * Ktot; idx += 256) {
        int r = idx / Ktot, k = idx - r * Ktot;
        int grow = m0 + r;
        float v = 0.f;
        if (grow < M) {
            if (k < K0) v = bfu2f(Hb[(long)grow * K0 + k]);
            else        v = Xf[(long)grow * K1 + (k - K0)];
        }
        As[r * AS_STRIDE + k] = f2bf(v);
    }

    v8f accS[4], accT[4];
    #pragma unroll
    for (int t = 0; t < 4; t++)
        #pragma unroll
        for (int v = 0; v < 8; v++) { accS[t][v] = 0.f; accT[t][v] = 0.f; }

    const int am = msub * 16 + (lane & 15);
    const int kh = lane >> 4;

    for (int kt = 0; kt < Ktot; kt += 32) {
        __syncthreads();
        for (int idx = tid; idx < NT * 32; idx += 256) {
            int n = idx >> 5, kk = idx & 31;
            long kg = kt + kk;
            Wss[n * WS_STRIDE + kk] = f2bf(Wsig[kg * NT + n]);
            Wts[n * WS_STRIDE + kk] = f2bf(Wtan[kg * NT + n]);
        }
        __syncthreads();

        v16bf a;
        #pragma unroll
        for (int v = 0; v < 8; v++) {
            int kb = (v < 4 ? 2 * v : 16 + 2 * (v - 4)) + 8 * kh;
            a[2 * v]     = As[am * AS_STRIDE + kt + kb];
            a[2 * v + 1] = As[am * AS_STRIDE + kt + kb + 1];
        }
        #pragma unroll
        for (int t = 0; t < 4; t++) {
            int nn = nhalf * 64 + t * 16 + (lane & 15);
            v16bf bs, bt;
            #pragma unroll
            for (int v = 0; v < 8; v++) {
                int kk = (v < 4 ? 2 * v : 16 + 2 * (v - 4)) + 8 * kh;
                bs[2 * v]     = Wss[nn * WS_STRIDE + kk];
                bs[2 * v + 1] = Wss[nn * WS_STRIDE + kk + 1];
                bt[2 * v]     = Wts[nn * WS_STRIDE + kk];
                bt[2 * v + 1] = Wts[nn * WS_STRIDE + kk + 1];
            }
            accS[t] = __builtin_amdgcn_wmma_f32_16x16x32_bf16(
                false, a, false, bs, (short)0, accS[t], false, false);
            accT[t] = __builtin_amdgcn_wmma_f32_16x16x32_bf16(
                false, a, false, bt, (short)0, accT[t], false, false);
        }
    }

    #pragma unroll
    for (int t = 0; t < 4; t++) {
        int n = nhalf * 64 + t * 16 + (lane & 15);
        float bsn = bsig[n], btn = btan[n];
        #pragma unroll
        for (int v = 0; v < 8; v++) {
            int m = msub * 16 + (lane >> 4) * 8 + v;
            int grow = m0 + m;
            if (grow < M) {
                float sx = accS[t][v] + bsn;
                sx = fminf(30.f, fmaxf(-30.f, sx));
                float g = (1.f / (1.f + __expf(sx))) * tanhf(accT[t][v] + btn);
                atomicAdd(&pooled[(long)batch[grow] * NT + n], g);
            }
        }
    }
}

// ---------------- tiny per-graph MLP head ----------------
__global__ __launch_bounds__(256)
void mlp_head(const float* __restrict__ pooled,
              const float* __restrict__ Wf1, const float* __restrict__ bf1,
              const float* __restrict__ Wf2, const float* __restrict__ bf2,
              const float* __restrict__ Wout, const float* __restrict__ bout,
              float* __restrict__ out) {
    __shared__ float p_s[128], f1_s[128], f2_s[258], red[256];
    const int g = blockIdx.x, tid = threadIdx.x;
    if (tid < 128) p_s[tid] = pooled[(long)g * 128 + tid];
    __syncthreads();
    if (tid < 128) {
        float s = bf1[tid];
        for (int k = 0; k < 128; k++) s += p_s[k] * Wf1[k * 128 + tid];
        f1_s[tid] = leaky(s);
    }
    __syncthreads();
    for (int j = tid; j < 258; j += 256) {
        float s = bf2[j];
        for (int k = 0; k < 128; k++) s += f1_s[k] * Wf2[k * 258 + j];
        f2_s[j] = leaky(s);
    }
    __syncthreads();
    float s = 0.f;
    for (int j = tid; j < 258; j += 256) s += f2_s[j] * Wout[j];
    red[tid] = s;
    __syncthreads();
    for (int off = 128; off > 0; off >>= 1) {
        if (tid < off) red[tid] += red[tid + off];
        __syncthreads();
    }
    if (tid == 0) out[g] = 1.f / (1.f + __expf(-(red[0] + bout[0])));
}

// ---------------- launcher ----------------
extern "C" void kernel_launch(void* const* d_in, const int* in_sizes, int n_in,
                              void* d_out, int out_size, void* d_ws, size_t ws_size,
                              hipStream_t stream) {
    const float* x     = (const float*)d_in[0];
    const int*   ei    = (const int*)d_in[1];
    const int*   batch = (const int*)d_in[2];
    const float* eattr = (const float*)d_in[3];
    const float* Wrel1 = (const float*)d_in[4];
    const float* brel1 = (const float*)d_in[5];
    const float* Wroot1= (const float*)d_in[6];
    const float* Wrel2 = (const float*)d_in[7];
    const float* brel2 = (const float*)d_in[8];
    const float* Wroot2= (const float*)d_in[9];
    const float* Wsig  = (const float*)d_in[10];
    const float* bsig  = (const float*)d_in[11];
    const float* Wtanh = (const float*)d_in[12];
    const float* btanh = (const float*)d_in[13];
    const float* Wf1   = (const float*)d_in[14];
    const float* bf1   = (const float*)d_in[15];
    const float* Wf2   = (const float*)d_in[16];
    const float* bf2   = (const float*)d_in[17];
    const float* Wout  = (const float*)d_in[18];
    const float* bout  = (const float*)d_in[19];
    const int* src = ei;
    const int* dst = ei + N_EDGES_C;
    float* out = (float*)d_out;

    // workspace carve-up
    char* ws = (char*)d_ws;
    float*          agg    = (float*)ws;                                   // 100000*128 f32 (51.2MB)
    unsigned short* h1     = (unsigned short*)(ws + (size_t)N_NODES_C * HID_C * 4);          // bf16 25.6MB
    unsigned short* h2     = (unsigned short*)(ws + (size_t)N_NODES_C * HID_C * 4
                                                  + (size_t)N_NODES_C * HID_C * 2);          // bf16 25.6MB
    float*          pooled = (float*)(ws + (size_t)N_NODES_C * HID_C * 4
                                         + (size_t)N_NODES_C * HID_C * 2 * 2);               // 256*128 f32

    const int ZB = 4096, SB = 8192;
    const int gemmBlocks = (N_NODES_C + MT - 1) / MT;   // 1563

    // conv1: agg1 = segsum(x[src]*w), h1 = leaky([agg1||x]@[Wrel1;Wroot1]+brel1)
    zero_f32<<<ZB, 256, 0, stream>>>(agg, (long)N_NODES_C * IN_DIM_C);
    scatter_f32<<<SB, 256, 0, stream>>>(x, src, dst, eattr, agg, 6 /*dim 64*/);
    gemm_bias_leaky<<<gemmBlocks, 256, 0, stream>>>(
        agg, nullptr, IN_DIM_C, x, nullptr, IN_DIM_C,
        Wrel1, Wroot1, IN_DIM_C, brel1, h1, N_NODES_C);

    // conv2: agg2 = segsum(h1[src]*w), h2 = leaky([agg2||h1]@[Wrel2;Wroot2]+brel2)
    zero_f32<<<ZB, 256, 0, stream>>>(agg, (long)N_NODES_C * HID_C);
    scatter_bf16<<<SB, 256, 0, stream>>>(h1, src, dst, eattr, agg, 7 /*dim 128*/);
    gemm_bias_leaky<<<gemmBlocks, 256, 0, stream>>>(
        agg, nullptr, HID_C, nullptr, h1, HID_C,
        Wrel2, Wroot2, HID_C, brel2, h2, N_NODES_C);

    // gate + pool: pooled[g] += sigmoid(-clip([h2||x]@Wsig+b)) * tanh([h2||x]@Wtanh+b)
    zero_f32<<<64, 256, 0, stream>>>(pooled, (long)N_GRAPHS_C * HID_C);
    gemm_gate_pool<<<gemmBlocks, 256, 0, stream>>>(
        h2, HID_C, x, IN_DIM_C, Wsig, bsig, Wtanh, btanh, batch, pooled, N_NODES_C);

    // head MLP -> sigmoid
    mlp_head<<<N_GRAPHS_C, 256, 0, stream>>>(pooled, Wf1, bf1, Wf2, bf2, Wout, bout, out);

    (void)in_sizes; (void)n_in; (void)out_size; (void)ws_size;
}
